// GraphAttentionLayer_6425271074940
// MI455X (gfx1250) — matrix operand order repaired and checked by reading, not compiled, not run on hardware
//
#include <hip/hip_runtime.h>

// ---------------------------------------------------------------------------
// GAT layer on MI455X (gfx1250): bf16 WMMA GEMMs + flash-style attention.
// Shapes: B=8, S=2048, D=768.
// 128x128 block tiles, 8 waves (wave32), each wave = 4x2 WMMA tiles/K-step.
// Software pipeline: register prefetch + double-buffered LDS (1 barrier/step).
// ---------------------------------------------------------------------------

typedef __bf16 bf16_t;
typedef __attribute__((ext_vector_type(16))) __bf16 v16bf;
typedef __attribute__((ext_vector_type(8)))  float  v8f;

#define BATCH 8
#define SEQ   2048
#define DIM   768
#define MROWS (BATCH * SEQ)      // 16384 flattened rows
#define KFUSE (2 * DIM)          // 1536
#define NEG_SLOPE 0.2f
#define LTILE (128 * 32)         // LDS tile elements

static __device__ __forceinline__ v8f wmma_bf16(v16bf a, v16bf b, v8f c) {
  // 8 args: (neg_a, A, neg_b, B, c_mod, C, reuse_a, reuse_b)
  return __builtin_amdgcn_wmma_f32_16x16x32_bf16(false, a, false, b, (short)0, c,
                                                 false, false);
}

// Pack two f32 -> one dword of bf16 pair (v_cvt_pk_bf16_f32) and store b32.
static __device__ __forceinline__ void st_pk(bf16_t* dst, float lo, float hi) {
  union { unsigned u; __bf16 h[2]; } p;
  p.h[0] = (__bf16)lo;
  p.h[1] = (__bf16)hi;
  *(unsigned*)dst = p.u;
}
static __device__ __forceinline__ void st_pk_raw(bf16_t* dst, bf16_t lo, bf16_t hi) {
  union { unsigned u; __bf16 h[2]; } p;
  p.h[0] = lo;
  p.h[1] = hi;
  *(unsigned*)dst = p.u;
}

// A fragment: 16x32 bf16 (lane<16: v0..7 K=0..15;
// lane>=16: K=16..31).  lA is row-major [128][32].
static __device__ __forceinline__ void load_a_frag(const bf16_t* lA, int mt,
                                                   int lane, v16bf& a) {
  const int row = mt * 16 + (lane & 15);
  const int kb  = (lane >> 4) * 16;
  const bf16_t* p = lA + row * 32;
#pragma unroll
  for (int r = 0; r < 8; ++r) {
    const int k = 2 * r + kb;
    a[2 * r]     = p[k];
    a[2 * r + 1] = p[k + 1];
  }
}

// B fragment: 32x16 bf16; lanes 0-15 hold K=0..15 of their column in v0..7,
// lanes 16-31 hold K=16..31.  lB stored transposed [n][k] so each lane's
// K-pairs are contiguous 4-byte LDS reads (lowers to ds_load_b128).
static __device__ __forceinline__ void load_b_frag(const bf16_t* lB, int nt,
                                                   int lane, v16bf& b) {
  const int col = nt * 16 + (lane & 15);
  const int kb  = (lane >> 4) * 16;
  const bf16_t* p = lB + col * 32 + kb;
#pragma unroll
  for (int r = 0; r < 8; ++r) {
    b[2 * r]     = p[2 * r];
    b[2 * r + 1] = p[2 * r + 1];
  }
}

// One K-step: 3 fragment streams -> 8 WMMAs (wave covers 64 rows x 32 cols).
static __device__ __forceinline__ void wmma_step(const bf16_t* lA, const bf16_t* lB,
                                                 int mtb, int ntb, int lane,
                                                 v8f acc[4][2]) {
  v16bf bf0, bf1;
  load_b_frag(lB, ntb + 0, lane, bf0);
  load_b_frag(lB, ntb + 1, lane, bf1);
#pragma unroll
  for (int mi = 0; mi < 4; ++mi) {
    v16bf a;
    load_a_frag(lA, mtb + mi, lane, a);
    acc[mi][0] = wmma_bf16(a, bf0, acc[mi][0]);
    acc[mi][1] = wmma_bf16(a, bf1, acc[mi][1]);
  }
}

// ---------------------------------------------------------------------------
// Kernel 1: xp = hidden @ W_gat  (fp32 in, bf16 out).
// ---------------------------------------------------------------------------
__global__ __launch_bounds__(256) void k_gemm_xp(const float* __restrict__ A,
                                                 const float* __restrict__ W,
                                                 bf16_t* __restrict__ xp) {
  __shared__ bf16_t lA[2][LTILE];
  __shared__ bf16_t lB[2][LTILE];
  const int tid = threadIdx.x, lane = tid & 31, wave = tid >> 5;
  const int mtb = (wave >> 2) * 4, ntb = (wave & 3) * 2;
  const int bm = blockIdx.x * 128, bn = blockIdx.y * 128;

  float2 pa[8];
  float  pb0[8], pb1[8];
  auto fetch = [&](int k0) {
#pragma unroll
    for (int i = 0; i < 8; ++i) {
      const int p = tid + i * 256, row = p >> 4, k = (p & 15) * 2;
      pa[i] = *(const float2*)(A + (size_t)(bm + row) * DIM + k0 + k);
    }
#pragma unroll
    for (int i = 0; i < 8; ++i) {
      const int p = tid + i * 256, n = p & 127, k = (p >> 7) * 2;
      pb0[i] = W[(size_t)(k0 + k) * DIM + bn + n];
      pb1[i] = W[(size_t)(k0 + k + 1) * DIM + bn + n];
    }
  };
  auto stage = [&](int buf) {
#pragma unroll
    for (int i = 0; i < 8; ++i) {
      const int p = tid + i * 256, row = p >> 4, k = (p & 15) * 2;
      st_pk(&lA[buf][row * 32 + k], pa[i].x, pa[i].y);
    }
#pragma unroll
    for (int i = 0; i < 8; ++i) {
      const int p = tid + i * 256, n = p & 127, k = (p >> 7) * 2;
      st_pk(&lB[buf][n * 32 + k], pb0[i], pb1[i]);
    }
  };

  v8f acc[4][2] = {};
  fetch(0);
  stage(0);
  __syncthreads();
  int cur = 0;
  for (int k0 = 0; k0 < DIM; k0 += 32) {
    const bool more = (k0 + 32) < DIM;
    if (more) fetch(k0 + 32);                     // global latency hidden by WMMA
    wmma_step(lA[cur], lB[cur], mtb, ntb, lane, acc);
    if (more) stage(cur ^ 1);                     // safe: last reads barrier-fenced
    __syncthreads();
    cur ^= 1;
  }

  const int col = lane & 15, rbase = (lane >> 4) * 8;
#pragma unroll
  for (int mi = 0; mi < 4; ++mi)
#pragma unroll
    for (int ni = 0; ni < 2; ++ni)
#pragma unroll
      for (int r = 0; r < 8; ++r) {
        const int grow = bm + (mtb + mi) * 16 + rbase + r;
        const int gcol = bn + (ntb + ni) * 16 + col;
        xp[(size_t)grow * DIM + gcol] = (bf16_t)acc[mi][ni][r];
      }
}

// ---------------------------------------------------------------------------
// Kernel 2: alpha_src/alpha_dst = xp . a_src / a_dst   (one block per row)
// ---------------------------------------------------------------------------
__global__ __launch_bounds__(256) void k_alpha(const bf16_t* __restrict__ xp,
                                               const float* __restrict__ a_src,
                                               const float* __restrict__ a_dst,
                                               float* __restrict__ asrc_o,
                                               float* __restrict__ adst_o) {
  const int row = blockIdx.x;
  float ps = 0.f, pd = 0.f;
  for (int k = threadIdx.x; k < DIM; k += 256) {
    const float v = (float)xp[(size_t)row * DIM + k];
    ps += v * a_src[k];
    pd += v * a_dst[k];
  }
#pragma unroll
  for (int o = 16; o > 0; o >>= 1) {
    ps += __shfl_xor(ps, o);
    pd += __shfl_xor(pd, o);
  }
  __shared__ float sS[8], sD[8];
  if ((threadIdx.x & 31) == 0) { sS[threadIdx.x >> 5] = ps; sD[threadIdx.x >> 5] = pd; }
  __syncthreads();
  if (threadIdx.x == 0) {
    float s = 0.f, d = 0.f;
#pragma unroll
    for (int w = 0; w < 8; ++w) { s += sS[w]; d += sD[w]; }
    asrc_o[row] = s;
    adst_o[row] = d;
  }
}

// ---------------------------------------------------------------------------
// Kernel 3: per-batch max of alpha_src (LeakyReLU monotone => row max of the
// logits is LeakyReLU(ad[i] + max_j as[j])).
// ---------------------------------------------------------------------------
__global__ __launch_bounds__(256) void k_bmax(const float* __restrict__ asrc,
                                              float* __restrict__ bmax) {
  const int b = blockIdx.x;
  float m = -3.4e38f;
  for (int j = threadIdx.x; j < SEQ; j += 256)
    m = fmaxf(m, asrc[b * SEQ + j]);
#pragma unroll
  for (int o = 16; o > 0; o >>= 1) m = fmaxf(m, __shfl_xor(m, o));
  __shared__ float sM[8];
  if ((threadIdx.x & 31) == 0) sM[threadIdx.x >> 5] = m;
  __syncthreads();
  if (threadIdx.x == 0) {
    float r = sM[0];
#pragma unroll
    for (int w = 1; w < 8; ++w) r = fmaxf(r, sM[w]);
    bmax[b] = r;
  }
}

// ---------------------------------------------------------------------------
// Kernel 4: per-row softmax stats: m_i and 1/denom_i  (one block per row)
// ---------------------------------------------------------------------------
__global__ __launch_bounds__(256) void k_stats(const float* __restrict__ asrc,
                                               const float* __restrict__ adst,
                                               const float* __restrict__ bmax,
                                               float* __restrict__ rmax,
                                               float* __restrict__ rinv) {
  const int row = blockIdx.x;
  const int b = row >> 11;  // row / SEQ
  const float ad = adst[row];
  float m = ad + bmax[b];
  m = m > 0.f ? m : NEG_SLOPE * m;
  float s = 0.f;
  for (int j = threadIdx.x; j < SEQ; j += 256) {
    float e = ad + asrc[b * SEQ + j];
    e = e > 0.f ? e : NEG_SLOPE * e;
    s += __expf(e - m);
  }
#pragma unroll
  for (int o = 16; o > 0; o >>= 1) s += __shfl_xor(s, o);
  __shared__ float sS[8];
  if ((threadIdx.x & 31) == 0) sS[threadIdx.x >> 5] = s;
  __syncthreads();
  if (threadIdx.x == 0) {
    float t = 0.f;
#pragma unroll
    for (int w = 0; w < 8; ++w) t += sS[w];
    rmax[row] = m;
    rinv[row] = 1.f / t;
  }
}

// ---------------------------------------------------------------------------
// Kernel 5: graph_out = attn @ xp + b_gat per batch — attention weights
// generated on the fly into LDS (never hit HBM), consumed by bf16 WMMA.
// M=2048, N=768, K=2048 per batch.  The exp/LeakyReLU generation co-executes
// with the WMMA pipe (XDL ops are tracked as TRANS).
// ---------------------------------------------------------------------------
__global__ __launch_bounds__(256) void k_attn(const bf16_t* __restrict__ xp,
                                              const float* __restrict__ adst,
                                              const float* __restrict__ asrc,
                                              const float* __restrict__ rmax,
                                              const float* __restrict__ rinv,
                                              const float* __restrict__ b_gat,
                                              bf16_t* __restrict__ gout) {
  __shared__ bf16_t lA[2][LTILE];
  __shared__ bf16_t lB[2][LTILE];
  const int tid = threadIdx.x, lane = tid & 31, wave = tid >> 5;
  const int mtb = (wave >> 2) * 4, ntb = (wave & 3) * 2;
  const int b = blockIdx.z;
  const int bm = blockIdx.x * 128, bn = blockIdx.y * 128;
  const size_t rowoff = (size_t)b * SEQ;

  // Each thread owns one attn row (fixed across K) for weight generation.
  const int srow = tid >> 1;        // 0..127
  const int skb  = (tid & 1) * 16;  // 0 or 16
  const float ad  = adst[rowoff + bm + srow];
  const float m   = rmax[rowoff + bm + srow];
  const float inv = rinv[rowoff + bm + srow];

  float  va[16];                    // 16 asrc values (float4 x4 loads)
  bf16_t pb0[8], pb1[8];
  auto fetch = [&](int k0) {
    const float4* ap = (const float4*)(asrc + rowoff + k0 + skb);
#pragma unroll
    for (int t = 0; t < 4; ++t) {
      const float4 v = ap[t];
      va[4 * t + 0] = v.x; va[4 * t + 1] = v.y;
      va[4 * t + 2] = v.z; va[4 * t + 3] = v.w;
    }
#pragma unroll
    for (int i = 0; i < 8; ++i) {
      const int p = tid + i * 256, n = p & 127, k = (p >> 7) * 2;
      pb0[i] = xp[(rowoff + k0 + k) * DIM + bn + n];
      pb1[i] = xp[(rowoff + k0 + k + 1) * DIM + bn + n];
    }
  };
  auto stage = [&](int buf) {
#pragma unroll
    for (int jp = 0; jp < 8; ++jp) {
      float e0 = ad + va[2 * jp];
      float e1 = ad + va[2 * jp + 1];
      e0 = e0 > 0.f ? e0 : NEG_SLOPE * e0;
      e1 = e1 > 0.f ? e1 : NEG_SLOPE * e1;
      st_pk(&lA[buf][srow * 32 + skb + 2 * jp],
            __expf(e0 - m) * inv, __expf(e1 - m) * inv);
    }
#pragma unroll
    for (int i = 0; i < 8; ++i) {
      const int p = tid + i * 256, n = p & 127, k = (p >> 7) * 2;
      st_pk_raw(&lB[buf][n * 32 + k], pb0[i], pb1[i]);
    }
  };

  v8f acc[4][2] = {};
  fetch(0);
  stage(0);
  __syncthreads();
  int cur = 0;
  for (int k0 = 0; k0 < SEQ; k0 += 32) {
    const bool more = (k0 + 32) < SEQ;
    if (more) fetch(k0 + 32);
    wmma_step(lA[cur], lB[cur], mtb, ntb, lane, acc);
    if (more) stage(cur ^ 1);
    __syncthreads();
    cur ^= 1;
  }

  const int col = lane & 15, rbase = (lane >> 4) * 8;
#pragma unroll
  for (int mi = 0; mi < 4; ++mi)
#pragma unroll
    for (int ni = 0; ni < 2; ++ni)
#pragma unroll
      for (int r = 0; r < 8; ++r) {
        const int grow = bm + (mtb + mi) * 16 + rbase + r;
        const int gcol = bn + (ntb + ni) * 16 + col;
        gout[(rowoff + grow) * DIM + gcol] = (bf16_t)(acc[mi][ni][r] + b_gat[gcol]);
      }
}

// ---------------------------------------------------------------------------
// Kernel 6: out = [transformer_output, graph_out] @ W_fus + b_fus.
// K=1536: first 768 from fp32 tout (cvt), second 768 from bf16 graph_out
// (raw dword copy, no conversion).  The K-segment choice is uniform per step.
// ---------------------------------------------------------------------------
__global__ __launch_bounds__(256) void k_fuse(const float* __restrict__ tout,
                                              const bf16_t* __restrict__ gout,
                                              const float* __restrict__ Wf,
                                              const float* __restrict__ bfus,
                                              float* __restrict__ out) {
  __shared__ bf16_t lA[2][LTILE];
  __shared__ bf16_t lB[2][LTILE];
  const int tid = threadIdx.x, lane = tid & 31, wave = tid >> 5;
  const int mtb = (wave >> 2) * 4, ntb = (wave & 3) * 2;
  const int bm = blockIdx.x * 128, bn = blockIdx.y * 128;

  float2 pa[8];
  float  pb0[8], pb1[8];
  auto fetch = [&](int k0) {
    if (k0 < DIM) {
#pragma unroll
      for (int i = 0; i < 8; ++i) {
        const int p = tid + i * 256, row = p >> 4, k = (p & 15) * 2;
        pa[i] = *(const float2*)(tout + (size_t)(bm + row) * DIM + k0 + k);
      }
    } else {
#pragma unroll
      for (int i = 0; i < 8; ++i) {
        const int p = tid + i * 256, row = p >> 4, k = (p & 15) * 2;
        const unsigned u =
            *(const unsigned*)(gout + (size_t)(bm + row) * DIM + (k0 - DIM) + k);
        pa[i].x = __uint_as_float(u);
      }
    }
#pragma unroll
    for (int i = 0; i < 8; ++i) {
      const int p = tid + i * 256, n = p & 127, k = (p >> 7) * 2;
      pb0[i] = Wf[(size_t)(k0 + k) * DIM + bn + n];
      pb1[i] = Wf[(size_t)(k0 + k + 1) * DIM + bn + n];
    }
  };
  auto stage = [&](int buf, int k0) {
    if (k0 < DIM) {
#pragma unroll
      for (int i = 0; i < 8; ++i) {
        const int p = tid + i * 256, row = p >> 4, k = (p & 15) * 2;
        st_pk(&lA[buf][row * 32 + k], pa[i].x, pa[i].y);
      }
    } else {
#pragma unroll
      for (int i = 0; i < 8; ++i) {
        const int p = tid + i * 256, row = p >> 4, k = (p & 15) * 2;
        *(unsigned*)(&lA[buf][row * 32 + k]) = __float_as_uint(pa[i].x);
      }
    }
#pragma unroll
    for (int i = 0; i < 8; ++i) {
      const int p = tid + i * 256, n = p & 127, k = (p >> 7) * 2;
      st_pk(&lB[buf][n * 32 + k], pb0[i], pb1[i]);
    }
  };

  v8f acc[4][2] = {};
  fetch(0);
  stage(0, 0);
  __syncthreads();
  int cur = 0;
  for (int k0 = 0; k0 < KFUSE; k0 += 32) {
    const bool more = (k0 + 32) < KFUSE;
    if (more) fetch(k0 + 32);
    wmma_step(lA[cur], lB[cur], mtb, ntb, lane, acc);
    if (more) stage(cur ^ 1, k0 + 32);
    __syncthreads();
    cur ^= 1;
  }

  const int col = lane & 15, rbase = (lane >> 4) * 8;
#pragma unroll
  for (int mi = 0; mi < 4; ++mi)
#pragma unroll
    for (int ni = 0; ni < 2; ++ni)
#pragma unroll
      for (int r = 0; r < 8; ++r) {
        const int grow = bm + (mtb + mi) * 16 + rbase + r;
        const int gcol = bn + (ntb + ni) * 16 + col;
        out[(size_t)grow * DIM + gcol] = acc[mi][ni][r] + bfus[gcol];
      }
}

// ---------------------------------------------------------------------------
// Host launcher
// ---------------------------------------------------------------------------
extern "C" void kernel_launch(void* const* d_in, const int* in_sizes, int n_in,
                              void* d_out, int out_size, void* d_ws, size_t ws_size,
                              hipStream_t stream) {
  (void)in_sizes; (void)n_in; (void)out_size; (void)ws_size;
  const float* hidden = (const float*)d_in[0];
  const float* tout   = (const float*)d_in[1];
  const float* W_gat  = (const float*)d_in[2];
  const float* a_src  = (const float*)d_in[3];
  const float* a_dst  = (const float*)d_in[4];
  const float* b_gat  = (const float*)d_in[5];
  const float* W_fus  = (const float*)d_in[6];
  const float* b_fus  = (const float*)d_in[7];
  float* out = (float*)d_out;

  // Workspace layout (bytes)
  char* ws = (char*)d_ws;
  const size_t XP_BYTES  = (size_t)MROWS * DIM * sizeof(bf16_t);  // 25,165,824
  const size_t VEC_BYTES = (size_t)MROWS * sizeof(float);         //     65,536
  bf16_t* xp     = (bf16_t*)(ws);
  bf16_t* gout   = (bf16_t*)(ws + XP_BYTES);
  float*  asrc_v = (float*)(ws + 2 * XP_BYTES);
  float*  adst_v = (float*)(ws + 2 * XP_BYTES + VEC_BYTES);
  float*  rmax_v = (float*)(ws + 2 * XP_BYTES + 2 * VEC_BYTES);
  float*  rinv_v = (float*)(ws + 2 * XP_BYTES + 3 * VEC_BYTES);
  float*  bmax_v = (float*)(ws + 2 * XP_BYTES + 4 * VEC_BYTES);

  dim3 blk(256);
  // 1) xp = hidden @ W_gat
  k_gemm_xp<<<dim3(MROWS / 128, DIM / 128), blk, 0, stream>>>(hidden, W_gat, xp);
  // 2) per-row alphas
  k_alpha<<<dim3(MROWS), blk, 0, stream>>>(xp, a_src, a_dst, asrc_v, adst_v);
  // 3) per-batch max of alpha_src
  k_bmax<<<dim3(BATCH), blk, 0, stream>>>(asrc_v, bmax_v);
  // 4) per-row softmax stats
  k_stats<<<dim3(MROWS), blk, 0, stream>>>(asrc_v, adst_v, bmax_v, rmax_v, rinv_v);
  // 5) graph_out = softmax(e) @ xp + b_gat (attn generated on the fly)
  k_attn<<<dim3(SEQ / 128, DIM / 128, BATCH), blk, 0, stream>>>(
      xp, adst_v, asrc_v, rmax_v, rinv_v, b_gat, gout);
  // 6) out = [tout, graph_out] @ W_fus + b_fus
  k_fuse<<<dim3(MROWS / 128, DIM / 128), blk, 0, stream>>>(tout, gout, W_fus, b_fus, out);
}
